// WordSAGE_50843822850677
// MI455X (gfx1250) — compile-verified, matrix operand order checked by reading, not run on hardware
//
#include <hip/hip_runtime.h>

// ---------------- types ----------------
typedef __attribute__((ext_vector_type(16))) __bf16 v16bf;
typedef __attribute__((ext_vector_type(8)))  float  v8f;

// bf16 helpers (raw ushort storage, RNE rounding)
static __device__ __forceinline__ unsigned short f2bf(float f) {
    union { float f; unsigned u; } x; x.f = f;
    unsigned r = x.u + 0x7FFFu + ((x.u >> 16) & 1u);
    return (unsigned short)(r >> 16);
}

// ---------------- constants ----------------
#define MROWS 16384
#define KG    2560   // padded 2500 (genes / hid / out feature dims)
#define KT    2688   // padded 2675 (train_feat dim)
#define NPAD  2560   // padded 2500 output columns
#define NCLSP 128    // padded 16 classifier columns
#define LP    40     // LDS row pitch in bf16 elements (32 + 8 pad)

// -------- CDNA5 async global->LDS path (guarded; falls back if absent) -----
#if defined(__AMDGCN__) && __has_builtin(__builtin_amdgcn_global_load_async_to_lds_b128)
#define USE_ASYNC 1
#else
#define USE_ASYNC 0
#endif

#if USE_ASYNC
// Exact parameter types per hipcc diagnostic:
//   param0: 'int __attribute__((vector_size(16))) __device__ *'  (global src)
//   param1: LDS destination
typedef int v4i_gcc __attribute__((__vector_size__(4 * sizeof(int))));
typedef __attribute__((address_space(1))) v4i_gcc as1_v4i;
typedef __attribute__((address_space(3))) v4i_gcc as3_v4i;

static __device__ __forceinline__ void async_cp16(const unsigned short* g,
                                                  unsigned short* l) {
    __builtin_amdgcn_global_load_async_to_lds_b128(
        (as1_v4i*)g, (as3_v4i*)l, 0, 0);
}

static __device__ __forceinline__ void wait_async0() {
#if __has_builtin(__builtin_amdgcn_s_wait_asynccnt)
    __builtin_amdgcn_s_wait_asynccnt(0);
#else
    asm volatile("s_wait_asynccnt 0x0" ::: "memory");
#endif
}
#endif

// ---------------- small utility kernels ----------------
__global__ void k_zero_f32(float* __restrict__ p, size_t n) {
    size_t i = (size_t)blockIdx.x * blockDim.x + threadIdx.x;
    if (i < n) p[i] = 0.0f;
}

__global__ void k_edges(const int* __restrict__ src, const int* __restrict__ dst,
                        float* __restrict__ A, float* __restrict__ deg, int E) {
    int e = blockIdx.x * blockDim.x + threadIdx.x;
    if (e >= E) return;
    int s = src[e], d = dst[e];
    atomicAdd(&A[(size_t)d * KG + s], 1.0f);
    atomicAdd(&deg[d], 1.0f);
}

// A_bf16[m][k] = bf16(A_f32[m][k] / max(deg[m],1))  -- folds mean into A
__global__ void k_scale_bf16(const float* __restrict__ A, const float* __restrict__ deg,
                             unsigned short* __restrict__ out) {
    size_t i = (size_t)blockIdx.x * blockDim.x + threadIdx.x;
    size_t n = (size_t)MROWS * KG;
    if (i >= n) return;
    int m = (int)(i / KG);
    float inv = 1.0f / fmaxf(deg[m], 1.0f);
    out[i] = f2bf(A[i] * inv);
}

// f32 -> padded bf16, optional transpose.  dst is [dstRows][dstPitch].
__global__ void k_convert_pad(const float* __restrict__ src, int srcRows, int srcCols,
                              unsigned short* __restrict__ dst, int dstRows, int dstPitch,
                              int transpose) {
    size_t i = (size_t)blockIdx.x * blockDim.x + threadIdx.x;
    size_t n = (size_t)dstRows * dstPitch;
    if (i >= n) return;
    int r = (int)(i / dstPitch), c = (int)(i % dstPitch);
    float v = 0.0f;
    if (!transpose) {
        if (r < srcRows && c < srcCols) v = src[(size_t)r * srcCols + c];
    } else {
        if (c < srcRows && r < srcCols) v = src[(size_t)c * srcCols + r];
    }
    dst[i] = f2bf(v);
}

__global__ void k_pad_f32(const float* __restrict__ src, int n,
                          float* __restrict__ dst, int npad) {
    int i = blockIdx.x * blockDim.x + threadIdx.x;
    if (i < npad) dst[i] = (i < n) ? src[i] : 0.0f;
}

// ---------------- WMMA fragment load from LDS ----------------
// Loads v16bf per ISA 16-bit A/B layout: elements kb..kb+7 then kb+16..kb+23.
static __device__ __forceinline__ v16bf ld_frag(const unsigned short* p) {
    union { uint4 u[2]; v16bf v; } x;
    x.u[0] = *reinterpret_cast<const uint4*>(p);
    x.u[1] = *reinterpret_cast<const uint4*>(p + 16);
    return x.v;
}

// ---------------- fused bf16 WMMA GEMM ----------------
// D[M][*] = act( X1[M][K1] @ W1[N][K1]^T + X2[M][K2] @ W2[N][K2]^T + bias )
// Block: 256 thr = 8 waves; block tile 128x128; wave tile 64(M)x32(N); K-step 32.
// All buffers padded: M % 128 == 0, N(padded) % 128 == 0, K % 32 == 0.
__global__ __launch_bounds__(256) void k_gemm(
    const unsigned short* __restrict__ X1, int ld1,
    const unsigned short* __restrict__ W1, int K1,
    const unsigned short* __restrict__ X2, int ld2,
    const unsigned short* __restrict__ W2, int K2,
    const float* __restrict__ bias, int relu,
    unsigned short* __restrict__ outBf, float* __restrict__ outF,
    int ldc, int Nstore)
{
    __shared__ unsigned short sA[2][128 * LP];
    __shared__ unsigned short sB[2][128 * LP];

    const int tid   = threadIdx.x;
    const int lane  = tid & 31;
    const int wave  = tid >> 5;
    const int waveM = wave & 1;     // 0..1 -> 64-row slab
    const int waveN = wave >> 1;    // 0..3 -> 32-col slab
    const int blockM = blockIdx.y;
    const int blockN = blockIdx.x;

    // global->LDS mapping: 128 rows x 32 bf16/row; 2 threads/row x 16 elems
    const int lrow  = tid & 127;
    const int le    = (tid >> 7) * 16;
    const int lofs  = lrow * LP + le;      // LDS element offset for this thread

    // fragment mapping (ISA 7.12.2, 16-bit 16x32 operand)
    const int fr = lane & 15;          // row (A) / col (B) within 16
    const int kb = (lane >> 4) * 8;    // K sub-base: 0 or 8

    v8f acc[4][2];
    for (int mt = 0; mt < 4; ++mt)
        for (int nt = 0; nt < 2; ++nt)
            for (int e = 0; e < 8; ++e) acc[mt][nt][e] = 0.0f;

    const int nc1 = K1 / 32;
    const int nc2 = K2 / 32;
    const int nct = nc1 + nc2;

    // global source addresses for chunk c (uniform pair selection per block)
    auto src_for = [&](int c, const unsigned short*& xp, const unsigned short*& wp) {
        if (c < nc1) {
            int k0 = c * 32;
            xp = X1 + (size_t)(blockM * 128 + lrow) * ld1 + k0 + le;
            wp = W1 + (size_t)(blockN * 128 + lrow) * ld1 + k0 + le;
        } else {
            int k0 = (c - nc1) * 32;
            xp = X2 + (size_t)(blockM * 128 + lrow) * ld2 + k0 + le;
            wp = W2 + (size_t)(blockN * 128 + lrow) * ld2 + k0 + le;
        }
    };

    auto compute = [&](int b) {
        v16bf af[4], bfr[2];
        for (int mt = 0; mt < 4; ++mt)
            af[mt] = ld_frag(&sA[b][(waveM * 64 + mt * 16 + fr) * LP + kb]);
        for (int nt = 0; nt < 2; ++nt)
            bfr[nt] = ld_frag(&sB[b][(waveN * 32 + nt * 16 + fr) * LP + kb]);
        for (int mt = 0; mt < 4; ++mt)
            for (int nt = 0; nt < 2; ++nt)
                acc[mt][nt] = __builtin_amdgcn_wmma_f32_16x16x32_bf16(
                    false, af[mt], false, bfr[nt],
                    (short)0, acc[mt][nt], false, false);
    };

#if USE_ASYNC
    // Double-buffered pipeline: DMA chunk c+1 overlaps WMMAs of chunk c.
    auto issue = [&](int c, int b) {
        const unsigned short *xp, *wp;
        src_for(c, xp, wp);
        async_cp16(xp,     &sA[b][lofs]);
        async_cp16(xp + 8, &sA[b][lofs + 8]);
        async_cp16(wp,     &sB[b][lofs]);
        async_cp16(wp + 8, &sB[b][lofs + 8]);
    };
    issue(0, 0);
    for (int c = 0; c < nct; ++c) {
        int cur = c & 1;
        wait_async0();          // my chunk-c DMAs landed in LDS
        __syncthreads();        // everyone's landed; prev reads of buf[cur^1] done
        if (c + 1 < nct) issue(c + 1, cur ^ 1);
        compute(cur);
    }
#else
    // Register-pipelined fallback: global loads for chunk c+1 issue before the
    // WMMAs of chunk c, so s_wait_loadcnt hides behind the compute phase.
    uint4 r0, r1, r2, r3;
    {
        const unsigned short *xp, *wp;
        src_for(0, xp, wp);
        r0 = *reinterpret_cast<const uint4*>(xp);
        r1 = *reinterpret_cast<const uint4*>(xp + 8);
        r2 = *reinterpret_cast<const uint4*>(wp);
        r3 = *reinterpret_cast<const uint4*>(wp + 8);
    }
    for (int c = 0; c < nct; ++c) {
        __syncthreads();        // prior reads of the buffer are complete
        *reinterpret_cast<uint4*>(&sA[0][lofs])     = r0;
        *reinterpret_cast<uint4*>(&sA[0][lofs + 8]) = r1;
        *reinterpret_cast<uint4*>(&sB[0][lofs])     = r2;
        *reinterpret_cast<uint4*>(&sB[0][lofs + 8]) = r3;
        __syncthreads();
        if (c + 1 < nct) {
            const unsigned short *xp, *wp;
            src_for(c + 1, xp, wp);
            r0 = *reinterpret_cast<const uint4*>(xp);
            r1 = *reinterpret_cast<const uint4*>(xp + 8);
            r2 = *reinterpret_cast<const uint4*>(wp);
            r3 = *reinterpret_cast<const uint4*>(wp + 8);
        }
        compute(0);
    }
#endif

    // ---------------- epilogue (branch-hoisted, branchless ReLU) -----------
    const int m0 = blockM * 128 + waveM * 64 + ((lane >> 4) * 8);
    const int n0 = blockN * 128 + waveN * 32 + fr;
    const float lo = relu ? 0.0f : -__builtin_inff();

    if (outF) {
        for (int nt = 0; nt < 2; ++nt) {
            int n = n0 + nt * 16;
            if (n < Nstore) {
                float bv = bias ? bias[n] : 0.0f;
                for (int mt = 0; mt < 4; ++mt) {
                    float* op = outF + (size_t)(m0 + mt * 16) * ldc + n;
                    for (int e = 0; e < 8; ++e) {
                        float v = fmaxf(acc[mt][nt][e] + bv, lo);
                        op[(size_t)e * ldc] = v;
                    }
                }
            }
        }
    } else {
        for (int nt = 0; nt < 2; ++nt) {
            int n = n0 + nt * 16;
            float bv = bias ? bias[n] : 0.0f;
            for (int mt = 0; mt < 4; ++mt) {
                unsigned short* op = outBf + (size_t)(m0 + mt * 16) * ldc + n;
                for (int e = 0; e < 8; ++e) {
                    float v = fmaxf(acc[mt][nt][e] + bv, lo);
                    op[(size_t)e * ldc] = f2bf(v);
                }
            }
        }
    }
}

// ---------------- launcher ----------------
extern "C" void kernel_launch(void* const* d_in, const int* in_sizes, int n_in,
                              void* d_out, int out_size, void* d_ws, size_t ws_size,
                              hipStream_t stream) {
    const float* gene_feat  = (const float*)d_in[0];   // [2500][2500]
    const float* train_feat = (const float*)d_in[1];   // [16384][2675]
    const int*   edge_src   = (const int*)d_in[2];
    const int*   edge_dst   = (const int*)d_in[3];
    const float* W1_self    = (const float*)d_in[4];   // [2500][2675]
    const float* W1_neigh   = (const float*)d_in[5];   // [2500][2500]
    const float* b1         = (const float*)d_in[6];
    const float* W2_self    = (const float*)d_in[7];
    const float* W2_neigh   = (const float*)d_in[8];
    const float* b2         = (const float*)d_in[9];
    const float* Wc1        = (const float*)d_in[10];
    const float* bc1        = (const float*)d_in[11];
    const float* Wc2        = (const float*)d_in[12];  // [16][2500]
    const float* bc2        = (const float*)d_in[13];
    const int E = in_sizes[2];

    // ---- workspace layout ----
    char* w = (char*)d_ws;
    size_t off = 0;
    auto alloc = [&](size_t bytes) -> char* {
        char* p = w + off;
        off = (off + bytes + 255) & ~(size_t)255;
        return p;
    };
    float*          A32  = (float*)alloc((size_t)MROWS * KG * 4);
    float*          deg  = (float*)alloc((size_t)MROWS * 4);
    unsigned short* Abf  = (unsigned short*)alloc((size_t)MROWS * KG * 2);
    unsigned short* Msrc = (unsigned short*)alloc((size_t)MROWS * KG * 2);
    unsigned short* Xt   = (unsigned short*)alloc((size_t)MROWS * KT * 2);
    unsigned short* Gt   = (unsigned short*)alloc((size_t)NPAD * KG * 2);
    unsigned short* W1s  = (unsigned short*)alloc((size_t)NPAD * KT * 2);
    unsigned short* W1n  = (unsigned short*)alloc((size_t)NPAD * KG * 2);
    unsigned short* W2s  = (unsigned short*)alloc((size_t)NPAD * KG * 2);
    unsigned short* W2n  = (unsigned short*)alloc((size_t)NPAD * KG * 2);
    unsigned short* Wc1p = (unsigned short*)alloc((size_t)NPAD * KG * 2);
    unsigned short* Wc2p = (unsigned short*)alloc((size_t)NCLSP * KG * 2);
    float* b1p  = (float*)alloc((size_t)NPAD * 4);
    float* b2p  = (float*)alloc((size_t)NPAD * 4);
    float* bc1p = (float*)alloc((size_t)NPAD * 4);
    float* bc2p = (float*)alloc((size_t)NCLSP * 4);
    // aliases (regions no longer live by the time these are written):
    unsigned short* hbuf  = (unsigned short*)A32;                        // after A consumed
    unsigned short* h2buf = (unsigned short*)A32 + (size_t)MROWS * KG;   // second half
    unsigned short* tbuf  = Abf;                                         // after GEMM1

    const int T = 256;
    auto blks = [&](size_t n) { return (unsigned)((n + T - 1) / T); };

    // 1) zero A and deg
    k_zero_f32<<<blks((size_t)MROWS * KG), T, 0, stream>>>(A32, (size_t)MROWS * KG);
    k_zero_f32<<<blks(MROWS), T, 0, stream>>>(deg, MROWS);

    // 2) scatter edges
    k_edges<<<blks(E), T, 0, stream>>>(edge_src, edge_dst, A32, deg, E);

    // 3) A/deg -> bf16
    k_scale_bf16<<<blks((size_t)MROWS * KG), T, 0, stream>>>(A32, deg, Abf);

    // 4) convert + pad operands
    k_convert_pad<<<blks((size_t)MROWS * KT), T, 0, stream>>>(train_feat, MROWS, 2675, Xt, MROWS, KT, 0);
    k_convert_pad<<<blks((size_t)NPAD * KG), T, 0, stream>>>(gene_feat, 2500, 2500, Gt, NPAD, KG, 1);
    k_convert_pad<<<blks((size_t)NPAD * KT), T, 0, stream>>>(W1_self, 2500, 2675, W1s, NPAD, KT, 0);
    k_convert_pad<<<blks((size_t)NPAD * KG), T, 0, stream>>>(W1_neigh, 2500, 2500, W1n, NPAD, KG, 0);
    k_convert_pad<<<blks((size_t)NPAD * KG), T, 0, stream>>>(W2_self, 2500, 2500, W2s, NPAD, KG, 0);
    k_convert_pad<<<blks((size_t)NPAD * KG), T, 0, stream>>>(W2_neigh, 2500, 2500, W2n, NPAD, KG, 0);
    k_convert_pad<<<blks((size_t)NPAD * KG), T, 0, stream>>>(Wc1, 2500, 2500, Wc1p, NPAD, KG, 0);
    k_convert_pad<<<blks((size_t)NCLSP * KG), T, 0, stream>>>(Wc2, 16, 2500, Wc2p, NCLSP, KG, 0);
    k_pad_f32<<<blks(NPAD), T, 0, stream>>>(b1, 2500, b1p, NPAD);
    k_pad_f32<<<blks(NPAD), T, 0, stream>>>(b2, 2500, b2p, NPAD);
    k_pad_f32<<<blks(NPAD), T, 0, stream>>>(bc1, 2500, bc1p, NPAD);
    k_pad_f32<<<blks(NCLSP), T, 0, stream>>>(bc2, 16, bc2p, NCLSP);

    dim3 gFull(NPAD / 128, MROWS / 128);   // (20, 128)
    dim3 gCls(NCLSP / 128, MROWS / 128);   // (1, 128)

    // 5) mean_src = Abf @ Gt^T
    k_gemm<<<gFull, T, 0, stream>>>(Abf, KG, Gt, KG,
                                    nullptr, 0, nullptr, 0,
                                    nullptr, 0, Msrc, nullptr, KG, NPAD);
    // 6) h = relu(Xt@W1s^T + Msrc@W1n^T + b1)
    k_gemm<<<gFull, T, 0, stream>>>(Xt, KT, W1s, KT,
                                    Msrc, KG, W1n, KG,
                                    b1p, 1, hbuf, nullptr, KG, NPAD);
    // 7) h2 = relu(h@W2s^T + Msrc@W2n^T + b2)
    k_gemm<<<gFull, T, 0, stream>>>(hbuf, KG, W2s, KG,
                                    Msrc, KG, W2n, KG,
                                    b2p, 1, h2buf, nullptr, KG, NPAD);
    // 8) t = relu(h2@Wc1^T + bc1)
    k_gemm<<<gFull, T, 0, stream>>>(h2buf, KG, Wc1p, KG,
                                    nullptr, 0, nullptr, 0,
                                    bc1p, 1, tbuf, nullptr, KG, NPAD);
    // 9) out = t@Wc2^T + bc2   (store only first 16 cols, f32)
    k_gemm<<<gCls, T, 0, stream>>>(tbuf, KG, Wc2p, KG,
                                   nullptr, 0, nullptr, 0,
                                   bc2p, 0, nullptr, (float*)d_out, 16, 16);
}
